// LSTM_28544352649958
// MI455X (gfx1250) — matrix operand order repaired and checked by reading, not compiled, not run on hardware
//
#include <hip/hip_runtime.h>
#include <cstdint>
#include <cstddef>

// ---------------------------------------------------------------------------
// Types for CDNA5 WMMA (wave32): v_wmma_f32_16x16x32_bf16
// ---------------------------------------------------------------------------
typedef __bf16 bf16_t;
typedef __attribute__((ext_vector_type(16))) __bf16       v16bf;
typedef __attribute__((ext_vector_type(8)))  float        v8f;
typedef __attribute__((ext_vector_type(4)))  unsigned int v4u;

#define L_T 512
#define L_B 32
#define L_I 512
#define L_H 1024
#define L_G (4 * L_H)       // 4096
#define L_TB (L_T * L_B)    // 16384
#define L_BH (L_B * L_H)    // 32768
#define L_TBH (L_T * L_B * L_H)

union AFrag { v16bf v; v4u q[2]; };

// A fragment (16x32, M x K), bf16, row-major source with leading dim `ld`.
// Lane l<16: row l, K = k0+0..15 contiguous.
// Lane l>=16: row l-16, K = k0+16..31 contiguous.
__device__ __forceinline__ v16bf load_a_frag(const bf16_t* A, int ld, int m0,
                                             int k0, int lane) {
  const int lr = lane & 15, hi = lane >> 4;
  const bf16_t* p = A + (size_t)(m0 + lr) * ld + k0 + hi * 16;
  AFrag f;
  f.q[0] = *(const v4u*)(p);
  f.q[1] = *(const v4u*)(p + 8);
  return f.v;
}

// B fragment (32x16, K x N) where B[k][n] = W[n0+n][k0+k], W row-major (ld=K).
// Lane l<16: col l, K = k0+0..15 contiguous; lane l>=16: col l-16, K = k0+16..31.
__device__ __forceinline__ v16bf load_b_frag(const bf16_t* W, int ld, int n0,
                                             int k0, int lane) {
  const int lr = lane & 15, hi = lane >> 4;
  const bf16_t* p = W + (size_t)(n0 + lr) * ld + k0 + hi * 16;
  return *(const v16bf*)(p);
}

__device__ __forceinline__ v8f wmma_bf16(v16bf a, v16bf b, v8f c) {
  return __builtin_amdgcn_wmma_f32_16x16x32_bf16(
      /*neg_a=*/false, a, /*neg_b=*/false, b,
      /*c_mod=*/(short)0, c, /*reuse_a=*/false, /*reuse_b=*/false);
}

// C/D tile layout: lane l<16 -> N=l, rows M=0..7 in acc[0..7];
//                  lane l>=16 -> N=l-16, rows M=8..15.
__device__ __forceinline__ v8f load_c_tile(const float* C, int ldc, int m0,
                                           int n0, int lane) {
  const int lr = lane & 15, mb = (lane >> 4) * 8;
  const float* p = C + (size_t)(m0 + mb) * ldc + n0 + lr;
  v8f acc;
#pragma unroll
  for (int r = 0; r < 8; ++r) acc[r] = p[(size_t)r * ldc];
  return acc;
}

__device__ __forceinline__ void store_d_tile(float* C, int ldc, int m0, int n0,
                                             int lane, v8f acc) {
  const int lr = lane & 15, mb = (lane >> 4) * 8;
  float* p = C + (size_t)(m0 + mb) * ldc + n0 + lr;
#pragma unroll
  for (int r = 0; r < 8; ++r) p[(size_t)r * ldc] = acc[r];
}

// ---------------------------------------------------------------------------
// Grid-wide barrier (monotonic ticket counter, agent scope).
// ---------------------------------------------------------------------------
__device__ __forceinline__ void grid_barrier(unsigned* cnt, unsigned nwg) {
  __threadfence();     // make this thread's stores device-visible
  __syncthreads();
  if (threadIdx.x == 0) {
    unsigned ticket = __hip_atomic_fetch_add(cnt, 1u, __ATOMIC_ACQ_REL,
                                             __HIP_MEMORY_SCOPE_AGENT);
    unsigned target = (ticket / nwg + 1u) * nwg;
    while (__hip_atomic_load(cnt, __ATOMIC_ACQUIRE,
                             __HIP_MEMORY_SCOPE_AGENT) < target) {
      __builtin_amdgcn_s_sleep(2);
    }
  }
  __syncthreads();
}

// ---------------------------------------------------------------------------
// fp32 -> bf16 conversion
// ---------------------------------------------------------------------------
__global__ __launch_bounds__(256) void cvt_bf16_kernel(
    const float* __restrict__ src, bf16_t* __restrict__ dst, size_t n) {
  size_t i = (size_t)blockIdx.x * blockDim.x + threadIdx.x;
  if (i < n) dst[i] = (bf16_t)src[i];
}

__global__ __launch_bounds__(256) void init_state_kernel(
    bf16_t* __restrict__ h, float* __restrict__ c, unsigned* __restrict__ cnt,
    int n) {
  int i = blockIdx.x * blockDim.x + threadIdx.x;
  if (i == 0) *cnt = 0u;
  if (i < n) { h[i] = (bf16_t)0.0f; c[i] = 0.0f; }
}

// ---------------------------------------------------------------------------
// Parallel GEMM:  C[m][n] = sum_k A[m][k] * W[n][k]   (bf16 in, fp32 out)
// block = 256 threads (8 waves); WG tile = 128(M) x 64(N).
// The 64x64 bf16 B panel is staged in LDS (double-buffered) so the 8 waves
// share one copy instead of 8 redundant L2 fetches. Row stride padded to 72
// elements (144 B): fragment readers hit banks r*36 mod 64 = 0,4,...,60,
// 4 consecutive banks each -> conflict-free ds_load_b128.
// grid = (N/64, M/128)
// ---------------------------------------------------------------------------
#define GK_STRIDE 72  // bf16 elements per LDS panel row (64 data + 8 pad)

__device__ __forceinline__ v16bf load_b_frag_lds(const bf16_t* sB, int jrow0,
                                                 int ks, int lane) {
  const int lr = lane & 15, hi = lane >> 4;
  const bf16_t* p = sB + (jrow0 + lr) * GK_STRIDE + ks + hi * 16;
  AFrag f;
  f.q[0] = *(const v4u*)(p);
  f.q[1] = *(const v4u*)(p + 8);
  return f.v;
}

__global__ __launch_bounds__(256) void gemm_bf16_kernel(
    const bf16_t* __restrict__ A, const bf16_t* __restrict__ W,
    float* __restrict__ C, int N, int K) {
  __shared__ __align__(16) bf16_t sB[2][64 * GK_STRIDE];

  const int tid = threadIdx.x;
  const int lane = tid & 31;
  const int wave = tid >> 5;
  const int n0 = blockIdx.x * 64;
  const int m0 = blockIdx.y * 128 + wave * 16;

  // Panel copy: 256 threads, each moves 32 B (row = tid/4, 2x16B segments).
  const int prow = tid >> 2;
  const int pcol = (tid & 3) * 16;
  const bf16_t* wrow = W + (size_t)(n0 + prow) * K + pcol;

  v8f zero = {};
  v8f acc[4] = {zero, zero, zero, zero};

  // Prologue: stage panel 0.
  {
    v4u p0 = *(const v4u*)(wrow);
    v4u p1 = *(const v4u*)(wrow + 8);
    bf16_t* d = &sB[0][prow * GK_STRIDE + pcol];
    *(v4u*)(d) = p0;
    *(v4u*)(d + 8) = p1;
  }
  __syncthreads();

  int cur = 0;
  for (int kb = 0; kb < K; kb += 64) {
    const bool more = (kb + 64) < K;
    v4u p0, p1;
    if (more) {  // prefetch next panel into registers while computing
      p0 = *(const v4u*)(wrow + kb + 64);
      p1 = *(const v4u*)(wrow + kb + 64 + 8);
    }
#pragma unroll
    for (int ks = 0; ks < 64; ks += 32) {
      v16bf a = load_a_frag(A, K, m0, kb + ks, lane);
#pragma unroll
      for (int j = 0; j < 4; ++j) {
        v16bf b = load_b_frag_lds(&sB[cur][0], j * 16, ks, lane);
        acc[j] = wmma_bf16(a, b, acc[j]);
      }
    }
    if (more) {
      // Writes go to the buffer whose last readers finished before the
      // barrier at the end of the previous iteration -> one barrier suffices.
      bf16_t* d = &sB[cur ^ 1][prow * GK_STRIDE + pcol];
      *(v4u*)(d) = p0;
      *(v4u*)(d + 8) = p1;
      __syncthreads();
      cur ^= 1;
    }
  }
#pragma unroll
  for (int j = 0; j < 4; ++j) store_d_tile(C, N, m0, n0 + j * 16, lane, acc[j]);
}

// ---------------------------------------------------------------------------
// Persistent LSTM layer kernel. 128 WGs x 128 threads = 512 wave32s;
// each wave owns one 16x16 tile of the (B=32) x (4H=4096) gate matrix.
// Per step:  gates = [xg_t or x_t@Wxh^T] + h@Whh^T  (WMMA, bf16)
//            barrier; elementwise LSTM cell update; barrier.
// Each B tile is shared by only 2 waves here (no LDS staging needed; Whh lives in L2).
// ---------------------------------------------------------------------------
__device__ __forceinline__ float sigmoidf_(float x) {
  return 1.0f / (1.0f + expf(-x));
}

__global__ __launch_bounds__(128) void lstm_layer_kernel(
    const float* __restrict__ xg,    // (T,B,G) precomputed x-projection or null
    const bf16_t* __restrict__ xin,  // (T,B,Kin) bf16 inputs (fused path)
    int Kin,
    const bf16_t* __restrict__ Wxh,  // (G,Kin) bf16 (fused path)
    const bf16_t* __restrict__ Whh,  // (G,H) bf16
    const float* __restrict__ bxh, const float* __restrict__ bhh,
    bf16_t* __restrict__ hstate,     // (B,H) bf16 running hidden
    float* __restrict__ cstate,      // (B,H) fp32 running cell
    float* __restrict__ gates,       // (B,G) fp32 scratch
    bf16_t* __restrict__ hs_bf,      // (T,B,H) bf16 hidden seq out (or null)
    float* __restrict__ out_f32,     // (T,B,H) fp32 hidden seq out (or null)
    float* __restrict__ hn, float* __restrict__ cn,  // (B,H) finals
    unsigned* __restrict__ counter) {
  const int lane = threadIdx.x & 31;
  const int wave = threadIdx.x >> 5;
  const int wid = blockIdx.x * 4 + wave;  // 0..511
  const int m0 = (wid & 1) * 16;          // batch-row tile (B=32 -> 2 tiles)
  const int n0 = (wid >> 1) * 16;         // gate-col tile (4096 -> 256 tiles)
  const unsigned nwg = gridDim.x;
  const int tid = blockIdx.x * blockDim.x + threadIdx.x;
  const int nthreads = gridDim.x * blockDim.x;

  for (int t = 0; t < L_T; ++t) {
    // ---- gate GEMM --------------------------------------------------------
    v8f acc;
    if (xg) {
      acc = load_c_tile(xg + (size_t)t * L_B * L_G, L_G, m0, n0, lane);
    } else {
      v8f zero = {};
      acc = zero;
      const bf16_t* xt = xin + (size_t)t * L_B * Kin;
      for (int k0 = 0; k0 < Kin; k0 += 32) {
        v16bf a = load_a_frag(xt, Kin, m0, k0, lane);
        v16bf b = load_b_frag(Wxh, Kin, n0, k0, lane);
        acc = wmma_bf16(a, b, acc);
      }
    }
    for (int k0 = 0; k0 < L_H; k0 += 32) {
      v16bf a = load_a_frag(hstate, L_H, m0, k0, lane);
      v16bf b = load_b_frag(Whh, L_H, n0, k0, lane);
      acc = wmma_bf16(a, b, acc);
    }
    store_d_tile(gates, L_G, m0, n0, lane, acc);

    grid_barrier(counter, nwg);

    // ---- elementwise cell update -----------------------------------------
    for (int e = tid; e < L_BH; e += nthreads) {
      int b = e >> 10;          // / H
      int j = e & (L_H - 1);    // % H
      const float* gb = gates + (size_t)b * L_G;
      float gi = gb[j]             + bxh[j]             + bhh[j];
      float gf = gb[L_H + j]       + bxh[L_H + j]       + bhh[L_H + j];
      float gg = gb[2 * L_H + j]   + bxh[2 * L_H + j]   + bhh[2 * L_H + j];
      float go = gb[3 * L_H + j]   + bxh[3 * L_H + j]   + bhh[3 * L_H + j];
      float cnew = sigmoidf_(gf) * cstate[e] + sigmoidf_(gi) * tanhf(gg);
      float hnew = sigmoidf_(go) * tanhf(cnew);
      cstate[e] = cnew;
      hstate[e] = (bf16_t)hnew;
      if (hs_bf)   hs_bf[(size_t)t * L_BH + e] = (bf16_t)hnew;
      if (out_f32) out_f32[(size_t)t * L_BH + e] = hnew;
      if (t == L_T - 1) { hn[e] = hnew; cn[e] = cnew; }
    }

    grid_barrier(counter, nwg);
  }
}

// ---------------------------------------------------------------------------
// Host launcher
// ---------------------------------------------------------------------------
extern "C" void kernel_launch(void* const* d_in, const int* in_sizes, int n_in,
                              void* d_out, int out_size, void* d_ws,
                              size_t ws_size, hipStream_t stream) {
  (void)in_sizes; (void)n_in; (void)out_size;
  const float* x    = (const float*)d_in[0];
  const float* Wxh0 = (const float*)d_in[1];
  const float* bxh0 = (const float*)d_in[2];
  const float* Whh0 = (const float*)d_in[3];
  const float* bhh0 = (const float*)d_in[4];
  const float* Wxh1 = (const float*)d_in[5];
  const float* bxh1 = (const float*)d_in[6];
  const float* Whh1 = (const float*)d_in[7];
  const float* bhh1 = (const float*)d_in[8];
  float* out = (float*)d_out;

  char* ws = (char*)d_ws;
  size_t off = 0;
  auto take = [&](size_t bytes) -> char* {
    char* p = ws + off;
    off = (off + bytes + 255) & ~(size_t)255;
    return p;
  };

  bf16_t* x_bf    = (bf16_t*)take((size_t)L_TB * L_I * 2);
  bf16_t* wxh0_bf = (bf16_t*)take((size_t)L_G * L_I * 2);
  bf16_t* whh0_bf = (bf16_t*)take((size_t)L_G * L_H * 2);
  bf16_t* wxh1_bf = (bf16_t*)take((size_t)L_G * L_H * 2);
  bf16_t* whh1_bf = (bf16_t*)take((size_t)L_G * L_H * 2);
  bf16_t* hs0_bf  = (bf16_t*)take((size_t)L_TBH * 2);
  bf16_t* hstate  = (bf16_t*)take((size_t)L_BH * 2);
  float*  cstate  = (float*)take((size_t)L_BH * 4);
  float*  gates   = (float*)take((size_t)L_B * L_G * 4);
  unsigned* counter = (unsigned*)take(256);
  size_t xg_bytes = (size_t)L_TB * L_G * 4;  // 268 MB
  bool use_xg = (off + xg_bytes) <= ws_size;
  float* xg = use_xg ? (float*)take(xg_bytes) : nullptr;

  auto cvt = [&](const float* src, bf16_t* dst, size_t n) {
    int blocks = (int)((n + 255) / 256);
    cvt_bf16_kernel<<<blocks, 256, 0, stream>>>(src, dst, n);
  };
  cvt(x,    x_bf,    (size_t)L_TB * L_I);
  cvt(Wxh0, wxh0_bf, (size_t)L_G * L_I);
  cvt(Whh0, whh0_bf, (size_t)L_G * L_H);
  cvt(Wxh1, wxh1_bf, (size_t)L_G * L_H);
  cvt(Whh1, whh1_bf, (size_t)L_G * L_H);

  float* out_hs = out;                       // (T,B,H)
  float* hn0 = out + (size_t)L_TBH;          // h_n layer0
  float* hn1 = hn0 + L_BH;                   // h_n layer1
  float* cn0 = hn1 + L_BH;                   // c_n layer0
  float* cn1 = cn0 + L_BH;                   // c_n layer1

  dim3 ggrid(L_G / 64, L_TB / 128);

  // ---- layer 0 ----
  if (use_xg)
    gemm_bf16_kernel<<<ggrid, 256, 0, stream>>>(x_bf, wxh0_bf, xg, L_G, L_I);
  init_state_kernel<<<(L_BH + 255) / 256, 256, 0, stream>>>(hstate, cstate,
                                                            counter, L_BH);
  lstm_layer_kernel<<<128, 128, 0, stream>>>(
      use_xg ? xg : nullptr, x_bf, L_I, wxh0_bf, whh0_bf, bxh0, bhh0, hstate,
      cstate, gates, hs0_bf, nullptr, hn0, cn0, counter);

  // ---- layer 1 ----
  if (use_xg)
    gemm_bf16_kernel<<<ggrid, 256, 0, stream>>>(hs0_bf, wxh1_bf, xg, L_G, L_H);
  init_state_kernel<<<(L_BH + 255) / 256, 256, 0, stream>>>(hstate, cstate,
                                                            counter, L_BH);
  lstm_layer_kernel<<<128, 128, 0, stream>>>(
      use_xg ? xg : nullptr, hs0_bf, L_H, wxh1_bf, whh1_bf, bxh1, bhh1, hstate,
      cstate, gates, nullptr, out_hs, hn1, cn1, counter);
}
